// SANetwork_55362128445564
// MI455X (gfx1250) — matrix operand order repaired and checked by reading, not compiled, not run on hardware
//
#include <hip/hip_runtime.h>
#include <hip/hip_bf16.h>
#include <stdint.h>

typedef __attribute__((ext_vector_type(16))) __bf16        v16bf;
typedef __attribute__((ext_vector_type(8)))  float         v8f;
typedef __attribute__((ext_vector_type(8)))  unsigned int  v8u;

#define S_LEN   200
#define S_PAD   208            // 13 tiles of 16
#define D_DIM   128
#define F_DIM   512
#define K_INT   4
#define ROW_B   272            // 128 bf16 (256B) + 16B pad -> conflict-free A reads
#define NEG_INF (-4294967295.0f)

static __device__ __forceinline__ unsigned short f2bf(float f) {
    union { float f; unsigned int u; } v; v.f = f;
    unsigned int u = v.u;
    u += 0x7FFFu + ((u >> 16) & 1u);     // round-to-nearest-even
    return (unsigned short)(u >> 16);
}
static __device__ __forceinline__ float bf2f(unsigned short h) {
    union { unsigned int u; float f; } v; v.u = ((unsigned int)h) << 16;
    return v.f;
}

// Hardware tanh: V_TANH_F32 is a CDNA5 TRANS op. Prefer the builtin; fall back
// to inline asm with a trailing v_nop to satisfy the TRANS-op result hazard.
static __device__ __forceinline__ float fast_tanh(float x) {
#if __has_builtin(__builtin_amdgcn_tanhf)
    return __builtin_amdgcn_tanhf(x);
#else
    float r;
    asm("v_tanh_f32 %0, %1\n\tv_nop" : "=v"(r) : "v"(x));
    return r;
#endif
}

// Pack W1 [128,512] fp32 -> bf16 B fragments in this kernel's contiguous k order (k = 16*half + s).
// dword index = ((nt*4 + kt)*32 + lane)*8 + j
//   lane 0-15 : N = nt*16 + lane,      K = kt*32 + 2j,2j+1
//   lane16-31 : N = nt*16 + lane-16,   K = kt*32 + 16 + 2j, +1
__global__ void pack_w1_kernel(const float* __restrict__ W1,
                               unsigned int* __restrict__ W1B) {
    int id   = blockIdx.x * blockDim.x + threadIdx.x;   // 0..32767
    int j    = id & 7;
    int lane = (id >> 3) & 31;
    int kt   = (id >> 8) & 3;
    int nt   = id >> 10;
    int n    = nt * 16 + (lane & 15);
    int k    = kt * 32 + ((lane >> 4) << 4) + 2 * j;
    unsigned int lo = f2bf(W1[(size_t)k * F_DIM + n]);
    unsigned int hi = f2bf(W1[(size_t)(k + 1) * F_DIM + n]);
    W1B[id] = lo | (hi << 16);
}

__global__ __launch_bounds__(256)
void sanet_fused_kernel(const float* __restrict__ X,        // [B,S,D]
                        const float* __restrict__ item_eb,  // [B,D]
                        const int*   __restrict__ mask,     // [B,S]
                        const unsigned int* __restrict__ W1B,
                        const float* __restrict__ b1,       // [512]
                        const float* __restrict__ W2,       // [512,4]
                        const float* __restrict__ b2,       // [4]
                        float*       __restrict__ out) {    // [B,D]
    __shared__ __attribute__((aligned(16))) unsigned char Xs[S_PAD * ROW_B];
    __shared__ float att[K_INT * S_PAD];      // logits, then softmax in place
    __shared__ float interest[K_INT * D_DIM];
    __shared__ float scores[K_INT];
    __shared__ int   best;

    const int b    = blockIdx.x;
    const int tid  = threadIdx.x;
    const int lane = tid & 31;
    const int wave = tid >> 5;

    // ---- Stage X[b] into LDS as bf16 (padded rows zeroed) ----
    const float* Xb = X + (size_t)b * S_LEN * D_DIM;
    for (int i = tid; i < S_LEN * (D_DIM / 2); i += 256) {   // pack 2 floats/iter
        int s = i >> 6, c = i & 63;
        float f0 = Xb[s * D_DIM + 2 * c];
        float f1 = Xb[s * D_DIM + 2 * c + 1];
        unsigned int p = (unsigned int)f2bf(f0) | ((unsigned int)f2bf(f1) << 16);
        *(unsigned int*)(Xs + s * ROW_B + c * 4) = p;
    }
    for (int i = tid; i < (S_PAD - S_LEN) * (D_DIM / 2); i += 256) {
        int s = S_LEN + (i >> 6), c = i & 63;
        *(unsigned int*)(Xs + s * ROW_B + c * 4) = 0u;
    }
    __syncthreads();

    // ---- GEMM1 (bf16 WMMA) fused with tanh and GEMM2 ----
    const int ah = lane >> 4;                 // which K-half this lane holds
    for (int mt = wave; mt < S_PAD / 16; mt += 8) {
        const unsigned char* arow = Xs + (mt * 16 + (lane & 15)) * ROW_B;
        v16bf afrag[4];
#pragma unroll
        for (int kt = 0; kt < 4; ++kt) {
            uint4 x0 = *(const uint4*)(arow + kt * 64 + ah * 32);
            uint4 x1 = *(const uint4*)(arow + kt * 64 + ah * 32 + 16);
            v8u t;
            t[0] = x0.x; t[1] = x0.y; t[2] = x0.z; t[3] = x0.w;
            t[4] = x1.x; t[5] = x1.y; t[6] = x1.z; t[7] = x1.w;
            afrag[kt] = __builtin_bit_cast(v16bf, t);
        }
        float acc[8][4];
#pragma unroll
        for (int r = 0; r < 8; ++r)
#pragma unroll
            for (int k = 0; k < 4; ++k) acc[r][k] = 0.0f;

        for (int nt = 0; nt < F_DIM / 16; ++nt) {
            v8f c = {0.f, 0.f, 0.f, 0.f, 0.f, 0.f, 0.f, 0.f};
#pragma unroll
            for (int kt = 0; kt < 4; ++kt) {
                v8u bv = *(const v8u*)(W1B + nt * 1024 + kt * 256 + lane * 8);
                v16bf bfrag = __builtin_bit_cast(v16bf, bv);
                c = __builtin_amdgcn_wmma_f32_16x16x32_bf16(
                        false, afrag[kt], false, bfrag,
                        (short)0, c, false, false);
            }
            // fuse: h = tanh(c + b1[f]); logits += h * W2[f,:]
            int f = nt * 16 + (lane & 15);
            float4 w2v = *(const float4*)(W2 + f * 4);
            float  b1f = b1[f];
#pragma unroll
            for (int r = 0; r < 8; ++r) {
                float h = fast_tanh(c[r] + b1f);
                acc[r][0] += h * w2v.x;
                acc[r][1] += h * w2v.y;
                acc[r][2] += h * w2v.z;
                acc[r][3] += h * w2v.w;
            }
        }
        // reduce across the 16 N-lanes of each half-wave
#pragma unroll
        for (int r = 0; r < 8; ++r)
#pragma unroll
            for (int k = 0; k < 4; ++k) {
                float v = acc[r][k];
                v += __shfl_xor(v, 1);
                v += __shfl_xor(v, 2);
                v += __shfl_xor(v, 4);
                v += __shfl_xor(v, 8);
                acc[r][k] = v;
            }
        if ((lane & 15) == 0) {
            int mbase = mt * 16 + 8 * ah;
#pragma unroll
            for (int r = 0; r < 8; ++r) {
                int s = mbase + r;
                if (s < S_LEN)
#pragma unroll
                    for (int k = 0; k < 4; ++k)
                        att[k * S_PAD + s] = acc[r][k] + b2[k];
            }
        }
    }
    __syncthreads();

    // ---- masked softmax over S, one wave per interest k ----
    if (wave < K_INT) {
        const int k = wave;
        const int* mrow = mask + (size_t)b * S_LEN;
        float mx = NEG_INF;
        for (int s = lane; s < S_LEN; s += 32) {
            float v = (mrow[s] != 0) ? att[k * S_PAD + s] : NEG_INF;
            att[k * S_PAD + s] = v;
            mx = fmaxf(mx, v);
        }
        for (int off = 16; off; off >>= 1) mx = fmaxf(mx, __shfl_xor(mx, off));
        float sum = 0.0f;
        for (int s = lane; s < S_LEN; s += 32) {
            float e = __expf(att[k * S_PAD + s] - mx);
            att[k * S_PAD + s] = e;
            sum += e;
        }
        for (int off = 16; off; off >>= 1) sum += __shfl_xor(sum, off);
        float inv = 1.0f / sum;
        for (int s = lane; s < S_LEN; s += 32) att[k * S_PAD + s] *= inv;
    }
    __syncthreads();

    // ---- interest[k,d] = sum_s att[k,s] * X[s,d] ----
    // one (k, d-pair) per thread: 4 * 64 = 256 units; packed bf16 dword reads
    {
        const int k  = tid >> 6;        // 0..3
        const int dp = tid & 63;        // d = 2*dp, 2*dp+1
        float a0 = 0.0f, a1 = 0.0f;
#pragma unroll 2
        for (int s = 0; s < S_LEN; ++s) {
            unsigned int p = *(const unsigned int*)(Xs + s * ROW_B + dp * 4);
            float w = att[k * S_PAD + s];
            a0 += w * bf2f((unsigned short)(p & 0xFFFFu));
            a1 += w * bf2f((unsigned short)(p >> 16));
        }
        interest[k * D_DIM + 2 * dp]     = a0;
        interest[k * D_DIM + 2 * dp + 1] = a1;
    }
    __syncthreads();

    // ---- scores, argmax (softmax is monotone -> argmax of raw scores) ----
    if (tid < K_INT) {
        const float* eb = item_eb + (size_t)b * D_DIM;
        float s = 0.0f;
        for (int d = 0; d < D_DIM; ++d) s += interest[tid * D_DIM + d] * eb[d];
        scores[tid] = s;
    }
    __syncthreads();
    if (tid == 0) {
        int   bi = 0;
        float bv = scores[0];
        for (int k = 1; k < K_INT; ++k)
            if (scores[k] > bv) { bv = scores[k]; bi = k; }
        best = bi;
    }
    __syncthreads();
    for (int d = tid; d < D_DIM; d += 256)
        out[(size_t)b * D_DIM + d] = interest[best * D_DIM + d];
}

extern "C" void kernel_launch(void* const* d_in, const int* in_sizes, int n_in,
                              void* d_out, int out_size, void* d_ws, size_t ws_size,
                              hipStream_t stream) {
    const float* item_his = (const float*)d_in[0];   // [B,200,128]
    const float* item_eb  = (const float*)d_in[1];   // [B,128]
    const int*   mask     = (const int*)d_in[2];     // [B,200]
    const float* W1       = (const float*)d_in[3];   // [128,512]
    const float* b1       = (const float*)d_in[4];   // [512]
    const float* W2       = (const float*)d_in[5];   // [512,4]
    const float* b2       = (const float*)d_in[6];   // [4]
    float*       out      = (float*)d_out;

    const int B = in_sizes[1] / D_DIM;               // 4096
    unsigned int* W1B = (unsigned int*)d_ws;         // 32768 dwords = 128 KB

    pack_w1_kernel<<<128, 256, 0, stream>>>(W1, W1B);
    sanet_fused_kernel<<<B, 256, 0, stream>>>(item_his, item_eb, mask,
                                              W1B, b1, W2, b2, out);
}
